// AdmissibleStatesHead_88416196755508
// MI455X (gfx1250) — compile-verified, matrix operand order checked by reading, not run on hardware
//
#include <hip/hip_runtime.h>
#include <hip/hip_bf16.h>
#include <cstdint>
#include <cstddef>

// ---------------------------------------------------------------------------
// AdmissibleStatesHead: fused  softmax(E @ W^T + b)  ->  6x8 digit marginals
//   E: [4096,1024] f32   W: [8192,1024] f32   b: [8192] f32   vs: [8192] i32
//   out: [6, 4096, 8] f32
// GEMM on v_wmma_f32_16x16x32_bf16 with bf16 hi/lo split (3 WMMAs per K-step
// ~ f32 accuracy). bf16 planes of W+E (48 MB) stay resident in the 192 MB L2.
// Each wave computes 2 M-tiles x 4 N-tiles; B fragments are shared by both
// M-tiles -> 24 WMMAs per (8 LDS + 16 global) b128 loads per k-step.
// A tiles are staged to LDS with GLOBAL_LOAD_ASYNC_TO_LDS_B128 (ASYNCcnt).
// ---------------------------------------------------------------------------

typedef __attribute__((ext_vector_type(16))) __bf16 v16bf;
typedef __attribute__((ext_vector_type(8)))  __bf16 v8bf;
typedef __attribute__((ext_vector_type(8)))  float  v8f;

#define B_ROWS   4096
#define N_VALID  8192
#define D_K      1024
#define WG_ROWS  32      // rows per workgroup (2 M-tiles)
#define N_CONC   6
#define NT       4       // N-tiles per wave (64 columns)
#define NM       2       // M-tiles per wave

// ---- LDS layout (dynamic, ~226 KB; WGP has 320 KB) -------------------------
#define OFF_AHI   0                        // 32*1024 bf16 = 65536 B
#define OFF_ALO   65536                    // 65536 B
#define OFF_PBUF  131072                   // 32*512 f32   = 65536 B
#define OFF_DIG   (OFF_PBUF + 65536)       // 8192 i32     = 32768 B
#define OFF_REDM  (OFF_DIG + 32768)        // 8*32 f32     = 1024 B
#define OFF_REDL  (OFF_REDM + 1024)        // 1024 B
#define OFF_RMAX  (OFF_REDL + 1024)        // 32 f32
#define OFF_RSUM  (OFF_RMAX + 128)         // 32 f32
#define LDS_BYTES (OFF_RSUM + 128)

// ---------------------------------------------------------------------------
// f32 -> (bf16 hi, bf16 lo) split:  x ~= hi + lo
// ---------------------------------------------------------------------------
__global__ void __launch_bounds__(256) split_bf16_kernel(
    const float* __restrict__ x, __bf16* __restrict__ hi,
    __bf16* __restrict__ lo, int n)
{
    int i = blockIdx.x * 256 + threadIdx.x;
    if (i < n) {
        float v = x[i];
        __bf16 h = (__bf16)v;
        hi[i] = h;
        lo[i] = (__bf16)(v - (float)h);
    }
}

// Async global->LDS 128-bit copy (GVS mode), tracked by ASYNCcnt.
static __device__ __forceinline__ void async_copy_b128(
    uint32_t lds_byte_off, uint32_t glb_byte_off, const void* sbase)
{
    asm volatile("global_load_async_to_lds_b128 %0, %1, %2"
                 :: "v"(lds_byte_off), "v"(glb_byte_off), "s"(sbase)
                 : "memory");
}

// Load one 16-bit WMMA fragment half-row: K = [0..7] and [16..23] relative
// to p (lanes 16-31 pass p offset by +8, per the ISA A/B layout tables).
__device__ __forceinline__ v16bf ld_frag16(const __bf16* p)
{
    v8bf a = *(const v8bf*)(p);
    v8bf b = *(const v8bf*)(p + 16);
    v16bf v;
#pragma unroll
    for (int i = 0; i < 8; ++i) { v[i] = a[i]; v[i + 8] = b[i]; }
    return v;
}

// 32x64 logits block: 2 M-tiles from LDS-staged A (32 x 1024 hi/lo),
// 4 N-tiles streamed from the L2-resident W hi/lo planes. B fragments are
// shared by both M-tiles; A fragments by all 4 N-tiles.
__device__ __forceinline__ void gemm_block(
    const __bf16* __restrict__ sAhi, const __bf16* __restrict__ sAlo,
    const __bf16* __restrict__ Whi,  const __bf16* __restrict__ Wlo,
    int n0, int r, int half, v8f acc[NM][NT])
{
    const __bf16* a0h = sAhi + (r)      * D_K;
    const __bf16* a0l = sAlo + (r)      * D_K;
    const __bf16* a1h = sAhi + (16 + r) * D_K;
    const __bf16* a1l = sAlo + (16 + r) * D_K;
    const __bf16* bh_row = Whi + (size_t)(n0 + r) * D_K;
    const __bf16* bl_row = Wlo + (size_t)(n0 + r) * D_K;
    const int ko = half * 8;
#pragma unroll 2
    for (int kk = 0; kk < D_K; kk += 32) {
        v16bf ah0 = ld_frag16(a0h + kk + ko);
        v16bf al0 = ld_frag16(a0l + kk + ko);
        v16bf ah1 = ld_frag16(a1h + kk + ko);
        v16bf al1 = ld_frag16(a1l + kk + ko);
#pragma unroll
        for (int t = 0; t < NT; ++t) {
            const size_t ts = (size_t)t * 16 * D_K;
            __builtin_prefetch(bh_row + ts + kk + 256, 0, 3);
            v16bf bh = ld_frag16(bh_row + ts + kk + ko);
            v16bf bl = ld_frag16(bl_row + ts + kk + ko);
            acc[0][t] = __builtin_amdgcn_wmma_f32_16x16x32_bf16(
                         false, ah0, false, bh, (short)0, acc[0][t], false, false);
            acc[1][t] = __builtin_amdgcn_wmma_f32_16x16x32_bf16(
                         false, ah1, false, bh, (short)0, acc[1][t], false, false);
            acc[0][t] = __builtin_amdgcn_wmma_f32_16x16x32_bf16(
                         false, ah0, false, bl, (short)0, acc[0][t], false, false);
            acc[1][t] = __builtin_amdgcn_wmma_f32_16x16x32_bf16(
                         false, ah1, false, bl, (short)0, acc[1][t], false, false);
            acc[0][t] = __builtin_amdgcn_wmma_f32_16x16x32_bf16(
                         false, al0, false, bh, (short)0, acc[0][t], false, false);
            acc[1][t] = __builtin_amdgcn_wmma_f32_16x16x32_bf16(
                         false, al1, false, bh, (short)0, acc[1][t], false, false);
        }
    }
}

// ---------------------------------------------------------------------------
// Fused kernel. One workgroup (8 wave32) per 32 embedding rows.
// Phase A: online per-row (max, sum-exp) over all 8192 states.
// Phase B: recompute tiles, accumulate digit marginals (no LDS atomics).
// ---------------------------------------------------------------------------
__global__ void __launch_bounds__(256) fused_marginals_kernel(
    const __bf16* __restrict__ Ehi, const __bf16* __restrict__ Elo,
    const __bf16* __restrict__ Whi, const __bf16* __restrict__ Wlo,
    const float* __restrict__ bias, const int* __restrict__ vstates,
    float* __restrict__ out)
{
    extern __shared__ char smem[];
    __bf16* sAhi  = (__bf16*)(smem + OFF_AHI);
    __bf16* sAlo  = (__bf16*)(smem + OFF_ALO);
    float*  pbuf  = (float*) (smem + OFF_PBUF);   // [32 rows][512 cols]
    int*    sdig  = (int*)   (smem + OFF_DIG);    // packed digits, all 8192
    float*  redM  = (float*) (smem + OFF_REDM);   // [8 waves][32 rows]
    float*  redL  = (float*) (smem + OFF_REDL);
    float*  rowMax= (float*) (smem + OFF_RMAX);   // [32]
    float*  rowSum= (float*) (smem + OFF_RSUM);   // [32]

    const int tid  = threadIdx.x;
    const int wave = tid >> 5;
    const int lane = tid & 31;
    const int half = lane >> 4;
    const int r    = lane & 15;
    const int wg   = blockIdx.x;            // 128 workgroups, 32 rows each

    // ---- stage A rows (hi/lo bf16) into LDS via async global->LDS copies --
    {
        const uint32_t ldsH = (uint32_t)(uintptr_t)sAhi;
        const uint32_t ldsL = (uint32_t)(uintptr_t)sAlo;
        const void* gh = (const void*)(Ehi + (size_t)wg * WG_ROWS * D_K);
        const void* gl = (const void*)(Elo + (size_t)wg * WG_ROWS * D_K);
#pragma unroll
        for (int i = 0; i < 16; ++i) {          // 4096 x 16B per plane
            uint32_t off = (uint32_t)(tid + 256 * i) * 16u;
            async_copy_b128(ldsH + off, off, gh);
            async_copy_b128(ldsL + off, off, gl);
        }
    }
    // ---- stage all packed state digits once (overlaps the async copies) ---
#pragma unroll
    for (int i = tid; i < N_VALID; i += 256) {
        int v = vstates[i];
        int dp = 0;
#pragma unroll
        for (int t = 0; t < N_CONC; ++t)
            dp |= ((v >> (15 - 3 * t)) & 7) << (3 * t);
        sdig[i] = dp;
    }
    asm volatile("s_wait_asynccnt 0" ::: "memory");
    __syncthreads();

    // ======================= Phase A: row max & sum-exp ====================
    float mx[NM][8], ls[NM][8];
#pragma unroll
    for (int mi = 0; mi < NM; ++mi)
#pragma unroll
        for (int j = 0; j < 8; ++j) { mx[mi][j] = -3.0e38f; ls[mi][j] = 0.f; }

    for (int c = 0; c < 16; ++c) {
        const int n0 = c * 512 + wave * 64;
        v8f acc[NM][NT] = {};
        gemm_block(sAhi, sAlo, Whi, Wlo, n0, r, half, acc);
#pragma unroll
        for (int t = 0; t < NT; ++t) {
            const float bsum = bias[n0 + t * 16 + r];
#pragma unroll
            for (int mi = 0; mi < NM; ++mi)
#pragma unroll
                for (int j = 0; j < 8; ++j) {
                    float x  = acc[mi][t][j] + bsum;
                    float mn = fmaxf(mx[mi][j], x);
                    ls[mi][j] = ls[mi][j] * __expf(mx[mi][j] - mn) + __expf(x - mn);
                    mx[mi][j] = mn;
                }
        }
    }
    // merge across the 16 lanes of each half-wave (same rows, different n)
#pragma unroll
    for (int mi = 0; mi < NM; ++mi)
#pragma unroll
        for (int j = 0; j < 8; ++j) {
#pragma unroll
            for (int m = 1; m < 16; m <<= 1) {
                float mo = __shfl_xor(mx[mi][j], m);
                float lo = __shfl_xor(ls[mi][j], m);
                float mn = fmaxf(mx[mi][j], mo);
                ls[mi][j] = ls[mi][j] * __expf(mx[mi][j] - mn) + lo * __expf(mo - mn);
                mx[mi][j] = mn;
            }
        }
    if (r == 0) {
#pragma unroll
        for (int mi = 0; mi < NM; ++mi)
#pragma unroll
            for (int j = 0; j < 8; ++j) {
                int m = mi * 16 + half * 8 + j;
                redM[wave * 32 + m] = mx[mi][j];
                redL[wave * 32 + m] = ls[mi][j];
            }
    }
    __syncthreads();
    if (tid < 32) {                          // merge the 8 waves
        float gm = -3.0e38f;
#pragma unroll
        for (int w = 0; w < 8; ++w) gm = fmaxf(gm, redM[w * 32 + tid]);
        float gl = 0.f;
#pragma unroll
        for (int w = 0; w < 8; ++w)
            gl += redL[w * 32 + tid] * __expf(redM[w * 32 + tid] - gm);
        rowMax[tid] = gm;
        rowSum[tid] = gl;
    }
    __syncthreads();

    // ======================= Phase B: digit marginals ======================
    // bin ownership: thread = group*32 + mrow owns bins b0..b0+5 of row mrow
    const int mrow  = tid & 31;
    const int group = tid >> 5;              // 0..7
    const int b0    = group * 6;             // 48 bins/row = 6 concepts x 8
    float bacc[6] = {0.f, 0.f, 0.f, 0.f, 0.f, 0.f};

    for (int c = 0; c < 16; ++c) {
        const int n0 = c * 512 + wave * 64;
        v8f acc[NM][NT] = {};
        gemm_block(sAhi, sAlo, Whi, Wlo, n0, r, half, acc);
#pragma unroll
        for (int t = 0; t < NT; ++t) {
            const float bsum = bias[n0 + t * 16 + r];
#pragma unroll
            for (int mi = 0; mi < NM; ++mi)
#pragma unroll
                for (int j = 0; j < 8; ++j) {
                    int m = mi * 16 + half * 8 + j;
                    pbuf[m * 512 + wave * 64 + t * 16 + r] =
                        __expf(acc[mi][t][j] + bsum - rowMax[m]);
                }
        }
        __syncthreads();
        const int* dbase = sdig + c * 512;
        for (int col = 0; col < 512; ++col) {
            float p  = pbuf[mrow * 512 + col];
            int   dp = dbase[col];
#pragma unroll
            for (int k = 0; k < 6; ++k) {
                int bb = b0 + k, t = bb >> 3, d = bb & 7;
                if (((dp >> (3 * t)) & 7) == d) bacc[k] += p;
            }
        }
        __syncthreads();
    }

    // normalize + store: out[t][wg*32+mrow][d]
    const float inv = 1.f / rowSum[mrow];
    const int   row = wg * WG_ROWS + mrow;
#pragma unroll
    for (int k = 0; k < 6; ++k) {
        int bb = b0 + k, t = bb >> 3, d = bb & 7;
        out[((size_t)t * B_ROWS + row) * 8 + d] = bacc[k] * inv;
    }
}

// ---------------------------------------------------------------------------
extern "C" void kernel_launch(void* const* d_in, const int* in_sizes, int n_in,
                              void* d_out, int out_size, void* d_ws, size_t ws_size,
                              hipStream_t stream)
{
    const float* E    = (const float*)d_in[0];   // [4096,1024]
    const float* W    = (const float*)d_in[1];   // [8192,1024]
    const float* bias = (const float*)d_in[2];   // [8192]
    const int*   vs   = (const int*)  d_in[3];   // [8192]
    float*       out  = (float*)d_out;

    // workspace: bf16 hi/lo planes (8+8+16+16 = 48 MB)
    __bf16* Ehi = (__bf16*)d_ws;
    __bf16* Elo = Ehi + (size_t)B_ROWS * D_K;
    __bf16* Whi = Elo + (size_t)B_ROWS * D_K;
    __bf16* Wlo = Whi + (size_t)N_VALID * D_K;

    const int nE = B_ROWS * D_K;
    const int nW = N_VALID * D_K;
    split_bf16_kernel<<<(nE + 255) / 256, 256, 0, stream>>>(E, Ehi, Elo, nE);
    split_bf16_kernel<<<(nW + 255) / 256, 256, 0, stream>>>(W, Whi, Wlo, nW);

    fused_marginals_kernel<<<B_ROWS / WG_ROWS, 256, LDS_BYTES, stream>>>(
        Ehi, Elo, Whi, Wlo, bias, vs, out);
}
